// KAN1_2491081031785
// MI455X (gfx1250) — matrix operand order repaired and checked by reading, not compiled, not run on hardware
//
#include <hip/hip_runtime.h>
#include <hip/hip_bf16.h>

// ---------------------------------------------------------------------------
// KAN layer on gfx1250: logits = [silu(x) | bsplines(x)] @ [Wbase | Wspline]^T
// then log_softmax. GEMM runs in bf16 WMMA (v_wmma_f32_16x16x32_bf16) with
// double-buffered GLOBAL_LOAD_ASYNC_TO_LDS_B128 tile staging (ASYNCcnt).
// K = 784 + 784*13 = 10976 = 343 * 32.
// ---------------------------------------------------------------------------

typedef __attribute__((ext_vector_type(16))) __bf16 v16bf;
typedef __attribute__((ext_vector_type(8)))  float  v8f;
typedef __attribute__((ext_vector_type(4)))  unsigned int u32x4;

#define BATCH   8192
#define IN_F    784
#define G_CNT   13            // GRID_SIZE + SPLINE_ORDER
#define K_TOT   10976         // 784 + 784*13, divisible by 32
#define N_OUT   1000
#define N_PAD   1024
#define BM      128
#define BN      128
#define BK      32
#define LDSR    40            // bf16 elems per LDS row (32 + 8 pad)

// round-to-nearest-even float -> bf16 bits
__device__ __forceinline__ unsigned short f2bf(float f) {
    unsigned int u = __float_as_uint(f);
    u += 0x7FFFu + ((u >> 16) & 1u);
    return (unsigned short)(u >> 16);
}

// async global -> LDS, 16 bytes per lane; tracked on ASYNCcnt
#define ASYNC_LOAD_B128(lds32, gaddr64)                                   \
    asm volatile("global_load_async_to_lds_b128 %0, %1, off"              \
                 :: "v"(lds32), "v"(gaddr64) : "memory")

// ---------------------------------------------------------------------------
// Kernel 1: pack combined weight matrix W[N_PAD][K_TOT] in bf16
// ---------------------------------------------------------------------------
__global__ __launch_bounds__(256) void kan_pack_w(
    const float* __restrict__ bw,   // (1000, 784)
    const float* __restrict__ sw,   // (1000, 784, 13)
    const float* __restrict__ sc,   // (1000, 784)
    unsigned short* __restrict__ Wbf)
{
    int idx = blockIdx.x * 256 + threadIdx.x;     // o * K_TOT + k
    int o = idx / K_TOT;
    int k = idx - o * K_TOT;
    float v = 0.0f;
    if (o < N_OUT) {
        if (k < IN_F) {
            v = bw[o * IN_F + k];
        } else {
            int r = k - IN_F;
            int i = r / G_CNT;
            int g = r - i * G_CNT;
            v = sw[(o * IN_F + i) * G_CNT + g] * sc[o * IN_F + i];
        }
    }
    Wbf[idx] = f2bf(v);
}

// ---------------------------------------------------------------------------
// Kernel 2: activations A[BATCH][K_TOT] = [silu(x) | bsplines(x)] in bf16
// ---------------------------------------------------------------------------
__global__ __launch_bounds__(256) void kan_activations(
    const float* __restrict__ x,          // (8192, 784) flat
    unsigned short* __restrict__ Abf)
{
    int idx = blockIdx.x * 256 + threadIdx.x;     // b * IN_F + i
    int b = idx / IN_F;
    int i = idx - b * IN_F;
    float xv = x[idx];

    // silu
    float sv = xv / (1.0f + __expf(-xv));

    // order-0 bases over 17 knots t_j = (j-3)*0.1, j = 0..16
    const float h = 0.1f;
    float bas[16];
#pragma unroll
    for (int j = 0; j < 16; ++j) {
        float t0 = (float)(j - 3) * h;
        float t1 = (float)(j - 2) * h;
        bas[j] = (xv >= t0 && xv < t1) ? 1.0f : 0.0f;
    }
    // Cox–de Boor, uniform grid: denominators are k*h
#pragma unroll
    for (int k = 1; k <= 3; ++k) {
        float inv = 1.0f / ((float)k * h);
#pragma unroll
        for (int j = 0; j < 16 - 1; ++j) {
            if (j < 16 - k) {
                float tj   = (float)(j - 3) * h;
                float tjk1 = (float)(j + k - 2) * h;   // t_{j+k+1} = (j+k+1-3)*h
                bas[j] = (xv - tj) * inv * bas[j] + (tjk1 - xv) * inv * bas[j + 1];
            }
        }
    }

    size_t row = (size_t)b * K_TOT;
    Abf[row + i] = f2bf(sv);
    size_t o = row + IN_F + (size_t)i * G_CNT;
#pragma unroll
    for (int g = 0; g < G_CNT; ++g) Abf[o + g] = f2bf(bas[g]);
}

// ---------------------------------------------------------------------------
// Kernel 3: bf16 WMMA GEMM: logits[BATCH][N_PAD] = A (MxK) * W^T (KxN)
// 256 threads = 8 waves in 2(M) x 4(N); wave tile 64x32 = 4x2 WMMA tiles.
// Tiles staged via double-buffered async global->LDS copies (ASYNCcnt).
// ---------------------------------------------------------------------------
struct Pair16 { u32x4 lo, hi; };

__global__ __launch_bounds__(256) void kan_gemm(
    const unsigned short* __restrict__ Abf,
    const unsigned short* __restrict__ Wbf,
    float* __restrict__ logits)
{
    __shared__ unsigned short sA[2][BM * LDSR];
    __shared__ unsigned short sB[2][BN * LDSR];

    int t    = threadIdx.x;
    int lane = t & 31;
    int wave = t >> 5;
    int wm   = wave >> 2;        // 0..1  (M direction)
    int wn   = wave & 3;         // 0..3  (N direction)
    int l15  = lane & 15;
    int kh   = lane >> 4;        // lane half selects K sub-range

    int mBase = blockIdx.x * BM;
    int nBase = blockIdx.y * BN;

    // tile-load mapping: 128 rows x 4 16B-chunks, 2 rows' worth per thread
    int rA = t >> 2;             // 0..63
    int cA = t & 3;              // chunk of 8 bf16

    // per-thread global base addresses (element k0 added per tile)
    const unsigned short* gA0 = Abf + (size_t)(mBase + rA)      * K_TOT + cA * 8;
    const unsigned short* gA1 = Abf + (size_t)(mBase + rA + 64) * K_TOT + cA * 8;
    const unsigned short* gB0 = Wbf + (size_t)(nBase + rA)      * K_TOT + cA * 8;
    const unsigned short* gB1 = Wbf + (size_t)(nBase + rA + 64) * K_TOT + cA * 8;

    // per-thread LDS destination byte addresses for both buffers
    // (generic LDS pointer truncated to 32 bits == wave-relative LDS offset)
    unsigned lA0[2], lA1[2], lB0[2], lB1[2];
#pragma unroll
    for (int bix = 0; bix < 2; ++bix) {
        lA0[bix] = (unsigned)(unsigned long long)(&sA[bix][(rA)      * LDSR + cA * 8]);
        lA1[bix] = (unsigned)(unsigned long long)(&sA[bix][(rA + 64) * LDSR + cA * 8]);
        lB0[bix] = (unsigned)(unsigned long long)(&sB[bix][(rA)      * LDSR + cA * 8]);
        lB1[bix] = (unsigned)(unsigned long long)(&sB[bix][(rA + 64) * LDSR + cA * 8]);
    }

    v8f acc[4][2] = {};

    const int nK = K_TOT / BK;   // 343

    // prologue: issue tile 0 into buffer 0
    {
        ASYNC_LOAD_B128(lA0[0], (unsigned long long)gA0);
        ASYNC_LOAD_B128(lA1[0], (unsigned long long)gA1);
        ASYNC_LOAD_B128(lB0[0], (unsigned long long)gB0);
        ASYNC_LOAD_B128(lB1[0], (unsigned long long)gB1);
    }

    for (int kt = 0; kt < nK; ++kt) {
        int cur = kt & 1;

        if (kt + 1 < nK) {
            // issue next tile into the other buffer, then wait only for the
            // 4 older async loads (this tile) -- next tile stays in flight
            size_t ko = (size_t)(kt + 1) * BK;
            ASYNC_LOAD_B128(lA0[cur ^ 1], (unsigned long long)(gA0 + ko));
            ASYNC_LOAD_B128(lA1[cur ^ 1], (unsigned long long)(gA1 + ko));
            ASYNC_LOAD_B128(lB0[cur ^ 1], (unsigned long long)(gB0 + ko));
            ASYNC_LOAD_B128(lB1[cur ^ 1], (unsigned long long)(gB1 + ko));
            asm volatile("s_wait_asynccnt 0x4" ::: "memory");
        } else {
            asm volatile("s_wait_asynccnt 0x0" ::: "memory");
        }
        __syncthreads();   // publish this tile's LDS data workgroup-wide

        const unsigned short* sAc = sA[cur];
        const unsigned short* sBc = sB[cur];

        // A fragments: 16x32 bf16; lanes 0-15 rows M, K {0..7,16..23};
        // lanes 16-31 same rows, K {8..15,24..31}.
        v16bf afr[4];
#pragma unroll
        for (int mt = 0; mt < 4; ++mt) {
            const unsigned short* p = &sAc[(wm * 64 + mt * 16 + l15) * LDSR + kh * 8];
            Pair16 pr;
            pr.lo = *(const u32x4*)(p);        // K = kh*8 .. kh*8+7
            pr.hi = *(const u32x4*)(p + 16);   // K = 16 + kh*8 .. +7
            afr[mt] = __builtin_bit_cast(v16bf, pr);
        }
        // B fragments: 32x16 bf16; lanes 0-15 K {0..7,16..23}, lanes 16-31
        // K {8..15,24..31}; column = n, sB row-major [n][k].
        v16bf bfr[2];
#pragma unroll
        for (int nt = 0; nt < 2; ++nt) {
            const unsigned short* p = &sBc[(wn * 32 + nt * 16 + l15) * LDSR + kh * 8];
            Pair16 pr;
            pr.lo = *(const u32x4*)(p);        // K = kh*8 .. kh*8+7
            pr.hi = *(const u32x4*)(p + 16);   // K = 16 + kh*8 .. +7
            bfr[nt] = __builtin_bit_cast(v16bf, pr);
        }

#pragma unroll
        for (int mt = 0; mt < 4; ++mt)
#pragma unroll
            for (int nt = 0; nt < 2; ++nt)
                acc[mt][nt] = __builtin_amdgcn_wmma_f32_16x16x32_bf16(
                    false, afr[mt], false, bfr[nt],
                    (short)0, acc[mt][nt], false, false);

        __syncthreads();   // all waves done reading buf[cur]; safe to refill
    }

    // epilogue: C layout — VGPR r: lanes 0-15 -> M=r, lanes 16-31 -> M=8+r; N=lane&15
#pragma unroll
    for (int mt = 0; mt < 4; ++mt) {
        int row = mBase + wm * 64 + mt * 16 + kh * 8;
#pragma unroll
        for (int nt = 0; nt < 2; ++nt) {
            int col = nBase + wn * 32 + nt * 16 + l15;
            float* p = logits + (size_t)row * N_PAD + col;
#pragma unroll
            for (int r = 0; r < 8; ++r)
                p[(size_t)r * N_PAD] = acc[mt][nt][r];
        }
    }
}

// ---------------------------------------------------------------------------
// Kernel 4: row-wise log_softmax over 1000 valid columns
// ---------------------------------------------------------------------------
__global__ __launch_bounds__(256) void kan_logsoftmax(
    const float* __restrict__ logits, float* __restrict__ out)
{
    __shared__ float red[256];
    int b = blockIdx.x;
    int t = threadIdx.x;
    const float* row = logits + (size_t)b * N_PAD;

    float m = -3.0e38f;
    for (int o = t; o < N_OUT; o += 256) m = fmaxf(m, row[o]);
    red[t] = m; __syncthreads();
    for (int s = 128; s > 0; s >>= 1) {
        if (t < s) red[t] = fmaxf(red[t], red[t + s]);
        __syncthreads();
    }
    m = red[0]; __syncthreads();

    float sum = 0.0f;
    for (int o = t; o < N_OUT; o += 256) sum += __expf(row[o] - m);
    red[t] = sum; __syncthreads();
    for (int s = 128; s > 0; s >>= 1) {
        if (t < s) red[t] += red[t + s];
        __syncthreads();
    }
    float lse = m + __logf(red[0]);

    for (int o = t; o < N_OUT; o += 256)
        out[(size_t)b * N_OUT + o] = row[o] - lse;
}

// ---------------------------------------------------------------------------
extern "C" void kernel_launch(void* const* d_in, const int* in_sizes, int n_in,
                              void* d_out, int out_size, void* d_ws, size_t ws_size,
                              hipStream_t stream)
{
    (void)in_sizes; (void)n_in; (void)out_size; (void)ws_size;
    const float* x  = (const float*)d_in[0];
    const float* bw = (const float*)d_in[1];
    const float* sw = (const float*)d_in[2];
    const float* sc = (const float*)d_in[3];
    float* out = (float*)d_out;

    char* ws = (char*)d_ws;
    size_t szW  = (size_t)N_PAD * K_TOT * sizeof(unsigned short);   // ~22.5 MB
    size_t offA = (szW + 255) & ~(size_t)255;
    size_t szA  = (size_t)BATCH * K_TOT * sizeof(unsigned short);   // ~180 MB
    size_t offL = (offA + szA + 255) & ~(size_t)255;                // logits ~32 MB

    unsigned short* Wbf = (unsigned short*)(ws);
    unsigned short* Abf = (unsigned short*)(ws + offA);
    float*          lg  = (float*)(ws + offL);

    kan_pack_w<<<(N_PAD * K_TOT) / 256, 256, 0, stream>>>(bw, sw, sc, Wbf);
    kan_activations<<<(BATCH * IN_F) / 256, 256, 0, stream>>>(x, Abf);

    dim3 grid(BATCH / BM, N_PAD / BN);   // 64 x 8
    kan_gemm<<<grid, 256, 0, stream>>>(Abf, Wbf, lg);

    kan_logsoftmax<<<BATCH, 256, 0, stream>>>(lg, out);
}
